// GAT_Multi_heads_35527969472816
// MI455X (gfx1250) — compile-verified, matrix operand order checked by reading, not run on hardware
//
#include <hip/hip_runtime.h>
#include <hip/hip_bf16.h>
#include <cstddef>

// ---------------- problem constants (match reference) ----------------
#define NNODES 100000
#define EEDGES 1600000
#define ETOT   (EEDGES + NNODES)   // edges + self-loops
#define FEAT   128                 // IN == HEADS*HID == OUT == 128
#define NTILES 8                   // FEAT/16 n-tiles
#define KCHUNKS 4                  // FEAT/32 k-chunks
#define NEG_SLOPE 0.2f

typedef __attribute__((ext_vector_type(16))) _Float16 v16h;
typedef __attribute__((ext_vector_type(8)))  _Float16 v8h;
typedef __attribute__((ext_vector_type(4)))  _Float16 v4h;
typedef __attribute__((ext_vector_type(8)))  float    v8f;
typedef __attribute__((ext_vector_type(4)))  float    v4f;

// ---------------- helpers ----------------
__device__ __forceinline__ float lrelu(float a) {
    return a > 0.f ? a : NEG_SLOPE * a;
}

// atomic max on float via int/uint ordering trick (works for mixed signs,
// buffer initialized to -inf). Emits native global_atomic_max/min.
__device__ __forceinline__ void atomicMaxFloat(float* addr, float val) {
    if (val >= 0.f)
        atomicMax((int*)addr, __float_as_int(val));
    else
        atomicMin((unsigned int*)addr, __float_as_uint(val));
}

// ---------------- elementwise f32 -> f16 convert (vec4) ----------------
__global__ void k_cvt_f16(const float* __restrict__ src, _Float16* __restrict__ dst, int n4) {
    int i = blockIdx.x * blockDim.x + threadIdx.x;
    if (i >= n4) return;
    v4f v = reinterpret_cast<const v4f*>(src)[i];
    v4h h;
    h[0] = (_Float16)v[0]; h[1] = (_Float16)v[1];
    h[2] = (_Float16)v[2]; h[3] = (_Float16)v[3];
    reinterpret_cast<v4h*>(dst)[i] = h;
}

// ---------------- fill ----------------
__global__ void k_fill(float* __restrict__ p, float v, int n) {
    int i = blockIdx.x * blockDim.x + threadIdx.x;
    if (i < n) p[i] = v;
}

// ---------------- pack a 128x128 f32 weight matrix into WMMA B fragments --
// One wave per (nTile, kChunk) fragment; 8*4 = 32 waves total. Each lane
// gathers its 16 f16 values (column col, K per ISA 7.12.2 16-bit pattern)
// and stores them contiguously, so the GEMM reads B as 2x global_load_b128.
__global__ void k_pack_w(const float* __restrict__ W, v16h* __restrict__ Bp) {
    const int lane = threadIdx.x & 31;
    const int gw   = (blockIdx.x * blockDim.x + threadIdx.x) >> 5;  // 0..31
    if (gw >= NTILES * KCHUNKS) return;
    const int nTile = gw >> 2;          // 0..7
    const int kc    = gw & 3;           // 0..3
    const int col   = nTile * 16 + (lane & 15);
    const int hi    = (lane >> 4) ? 8 : 0;
    v16h b;
#pragma unroll
    for (int v = 0; v < 8; ++v) {
        int kk = kc * 32 + ((v & 4) << 2) + hi + ((v & 3) << 1);
        b[2 * v]     = (_Float16)W[(size_t)kk * FEAT + col];
        b[2 * v + 1] = (_Float16)W[(size_t)(kk + 1) * FEAT + col];
    }
    Bp[gw * 32 + lane] = b;
}

// ---------------- WMMA GEMM: C[M,128] = A[M,128] x W[128,128] -------------
// A: row-major f16. Bp: pre-packed B fragments [nTile][kChunk][lane][16].
// One 16x16 tile per wave (8 waves/block = full N), K chained in chunks of
// 32 through v_wmma_f32_16x16x32_f16 with f32 accumulation.
// Inner loop: 2x b128 A loads + 2x b128 B loads + 1 wmma.
__global__ void k_gemm_wmma(const _Float16* __restrict__ A,
                            const v16h* __restrict__ Bp,
                            float* __restrict__ C, int M) {
    const int lane = threadIdx.x & 31;
    const int wid  = threadIdx.x >> 5;      // 8 waves -> 8 n-tiles
    const int m0   = blockIdx.x * 16;
    const int r    = lane & 15;
    const int hi   = (lane >> 4) ? 8 : 0;   // lanes 16-31 hold K+8 (and M+8 of D)
    const int row  = m0 + r;
    const int col  = wid * 16 + r;

    // A fragment per lane = two contiguous 16B chunks: K = k0+hi+[0..7] and
    // K = k0+16+hi+[0..7] (ISA 7.12.2 16-bit A layout).
    const v8h* pa = reinterpret_cast<const v8h*>(A + (size_t)row * FEAT + hi);
    const v16h* pb = Bp + wid * KCHUNKS * 32 + lane;

    v8f c = {};
#pragma unroll
    for (int kc = 0; kc < KCHUNKS; ++kc) {
        v8h a0 = pa[kc * 4 + 0];
        v8h a1 = pa[kc * 4 + 2];
        v16h a = __builtin_shufflevector(a0, a1, 0, 1, 2, 3, 4, 5, 6, 7,
                                                 8, 9, 10, 11, 12, 13, 14, 15);
        v16h b = pb[kc * 32];
        c = __builtin_amdgcn_wmma_f32_16x16x32_f16(
                /*neg_a=*/false, a, /*neg_b=*/false, b,
                /*c_mod=*/(short)0, c, /*reuse_a=*/false, /*reuse_b=*/false);
    }
#pragma unroll
    for (int j = 0; j < 8; ++j) {
        int m = m0 + j + hi;                // C/D layout: VGPR j -> row j (+8 hi)
        C[(size_t)m * FEAT + col] = c[j];
    }
}

// ---------------- per-(node,head) attention coefficients ----------------
// a_s[n,h] = dot(xp[n, h*32 : h*32+32], att_src[h]);  same for a_d.
__global__ void k_attn(const float* __restrict__ xp,
                       const float* __restrict__ att_src,
                       const float* __restrict__ att_dst,
                       float* __restrict__ a_s, float* __restrict__ a_d, int nh) {
    int t = blockIdx.x * blockDim.x + threadIdx.x;
    if (t >= nh) return;
    int node = t >> 2, h = t & 3;
    const v4f* xv = reinterpret_cast<const v4f*>(xp + (size_t)node * FEAT + h * 32);
    const v4f* sv = reinterpret_cast<const v4f*>(att_src + h * 32);
    const v4f* dv = reinterpret_cast<const v4f*>(att_dst + h * 32);
    float ss = 0.f, sd = 0.f;
#pragma unroll
    for (int i = 0; i < 8; ++i) {
        v4f x = xv[i], s = sv[i], d = dv[i];
        ss += x[0]*s[0] + x[1]*s[1] + x[2]*s[2] + x[3]*s[3];
        sd += x[0]*d[0] + x[1]*d[1] + x[2]*d[2] + x[3]*d[3];
    }
    a_s[t] = ss;
    a_d[t] = sd;
}

// ---------------- edge pass 1: segment max (thread per edge) ------------
// idx >= EEDGES encodes the self-loop for node idx-EEDGES.
__global__ void k_edge_max(const int* __restrict__ src, const int* __restrict__ dst,
                           const float* __restrict__ a_s, const float* __restrict__ a_d,
                           float* __restrict__ mx) {
    int i = blockIdx.x * blockDim.x + threadIdx.x;
    if (i >= ETOT) return;
    int s, d;
    if (i < EEDGES) { s = src[i]; d = dst[i]; } else { s = d = i - EEDGES; }
    v4f as = reinterpret_cast<const v4f*>(a_s)[s];
    v4f ad = reinterpret_cast<const v4f*>(a_d)[d];
#pragma unroll
    for (int h = 0; h < 4; ++h)
        atomicMaxFloat(&mx[d * 4 + h], lrelu(as[h] + ad[h]));
}

// ---------------- edge pass 2: softmax denom + GCN degree ---------------
__global__ void k_edge_denom(const int* __restrict__ src, const int* __restrict__ dst,
                             const float* __restrict__ ew,
                             const float* __restrict__ a_s, const float* __restrict__ a_d,
                             const float* __restrict__ mx,
                             float* __restrict__ denom, float* __restrict__ deg) {
    int i = blockIdx.x * blockDim.x + threadIdx.x;
    if (i >= ETOT) return;
    int s, d; float w;
    if (i < EEDGES) { s = src[i]; d = dst[i]; w = ew[i]; }
    else            { s = d = i - EEDGES;     w = 1.0f;  }
    v4f as = reinterpret_cast<const v4f*>(a_s)[s];
    v4f ad = reinterpret_cast<const v4f*>(a_d)[d];
    v4f m  = reinterpret_cast<const v4f*>(mx)[d];
#pragma unroll
    for (int h = 0; h < 4; ++h)
        atomicAdd(&denom[d * 4 + h], __expf(lrelu(as[h] + ad[h]) - m[h]));
    atomicAdd(&deg[d], w);
}

// ---------------- edge pass 3: GAT weighted scatter (wave per edge) -----
// Lane l handles 4 contiguous floats of the 128-wide row; head = lane>>3.
__global__ void k_gat_scatter(const int* __restrict__ src, const int* __restrict__ dst,
                              const float* __restrict__ a_s, const float* __restrict__ a_d,
                              const float* __restrict__ mx, const float* __restrict__ denom,
                              const float* __restrict__ xp, float* __restrict__ y) {
    const int lane = threadIdx.x & 31;
    const int e    = blockIdx.x * (blockDim.x >> 5) + (threadIdx.x >> 5);
    if (e >= ETOT) return;
    int s, d;
    if (e < EEDGES) { s = src[e]; d = dst[e]; } else { s = d = e - EEDGES; }
    const int h = lane >> 3;
    float al = lrelu(a_s[s * 4 + h] + a_d[d * 4 + h]);
    float a  = __expf(al - mx[d * 4 + h]) / denom[d * 4 + h];
    v4f v = reinterpret_cast<const v4f*>(xp)[(size_t)s * 32 + lane];
    float* out = y + (size_t)d * FEAT + lane * 4;
    atomicAdd(out + 0, v[0] * a);
    atomicAdd(out + 1, v[1] * a);
    atomicAdd(out + 2, v[2] * a);
    atomicAdd(out + 3, v[3] * a);
}

// ---------------- dinv = deg>0 ? rsqrt(deg) : 0 (in place) --------------
__global__ void k_dinv(float* __restrict__ deg, int n) {
    int i = blockIdx.x * blockDim.x + threadIdx.x;
    if (i >= n) return;
    float dv = deg[i];
    deg[i] = dv > 0.f ? rsqrtf(dv) : 0.f;
}

// ---------------- edge pass 4: GCN normalized scatter (wave per edge) ---
__global__ void k_gcn_scatter(const int* __restrict__ src, const int* __restrict__ dst,
                              const float* __restrict__ ew, const float* __restrict__ dinv,
                              const float* __restrict__ hbuf, float* __restrict__ out) {
    const int lane = threadIdx.x & 31;
    const int e    = blockIdx.x * (blockDim.x >> 5) + (threadIdx.x >> 5);
    if (e >= ETOT) return;
    int s, d; float w;
    if (e < EEDGES) { s = src[e]; d = dst[e]; w = ew[e]; }
    else            { s = d = e - EEDGES;     w = 1.0f;  }
    float norm = dinv[s] * w * dinv[d];
    v4f v = reinterpret_cast<const v4f*>(hbuf)[(size_t)s * 32 + lane];
    float* o = out + (size_t)d * FEAT + lane * 4;
    atomicAdd(o + 0, v[0] * norm);
    atomicAdd(o + 1, v[1] * norm);
    atomicAdd(o + 2, v[2] * norm);
    atomicAdd(o + 3, v[3] * norm);
}

// ================= host-side launch =================
extern "C" void kernel_launch(void* const* d_in, const int* in_sizes, int n_in,
                              void* d_out, int out_size, void* d_ws, size_t ws_size,
                              hipStream_t stream) {
    (void)in_sizes; (void)n_in; (void)out_size; (void)ws_size;

    const float* x       = (const float*)d_in[0];          // [N,128]
    const int*   eidx    = (const int*)  d_in[1];          // [2,E]
    const int*   src     = eidx;
    const int*   dst     = eidx + EEDGES;
    const float* ew      = (const float*)d_in[2];          // [E]
    const float* W_gat   = (const float*)d_in[3];          // [128,128]
    const float* att_src = (const float*)d_in[4];          // [4,32]
    const float* att_dst = (const float*)d_in[5];          // [4,32]
    const float* W_gcn   = (const float*)d_in[6];          // [128,128]
    float*       out     = (float*)d_out;                  // [N,128]

    // ---- workspace carve (256B aligned), with aliasing for reuse ----
    char*  ws  = (char*)d_ws;
    size_t off = 0;
    auto alloc = [&](size_t bytes) -> void* {
        void* p = ws + off;
        off += (bytes + 255) & ~(size_t)255;
        return p;
    };
    float*    xp     = (float*)   alloc((size_t)NNODES * FEAT * 4);  // later reused as h
    _Float16* xh     = (_Float16*)alloc((size_t)NNODES * FEAT * 2);  // later reused as y_gat_h
    v16h*     WgatBp = (v16h*)    alloc((size_t)NTILES * KCHUNKS * 32 * 32);  // packed fragments
    v16h*     WgcnBp = (v16h*)    alloc((size_t)NTILES * KCHUNKS * 32 * 32);
    float*    a_s    = (float*)   alloc((size_t)NNODES * 4 * 4);
    float*    a_d    = (float*)   alloc((size_t)NNODES * 4 * 4);
    float*    mx     = (float*)   alloc((size_t)NNODES * 4 * 4);
    float*    denom  = (float*)   alloc((size_t)NNODES * 4 * 4);
    float*    deg    = (float*)   alloc((size_t)NNODES * 4);         // becomes dinv in place
    float*    y_gat  = (float*)   alloc((size_t)NNODES * FEAT * 4);
    float*    hbuf   = xp;        // h aliases xp (xp dead after GAT scatter)
    _Float16* ygh    = xh;        // y_gat f16 aliases xh (xh dead after GEMM1)

    const int BLK = 256;
    auto blks = [](long long n, int b) { return (unsigned)((n + b - 1) / b); };

    // 1) convert x to f16; pack both weight matrices into B fragments
    k_cvt_f16<<<blks((long long)NNODES * FEAT / 4, BLK), BLK, 0, stream>>>(x, xh, NNODES * FEAT / 4);
    k_pack_w<<<4, BLK, 0, stream>>>(W_gat, WgatBp);
    k_pack_w<<<4, BLK, 0, stream>>>(W_gcn, WgcnBp);

    // 2) xp = x @ W_gat   (WMMA; M=100000=6250*16)
    k_gemm_wmma<<<NNODES / 16, 256, 0, stream>>>(xh, WgatBp, xp, NNODES);

    // 3) attention coefficients
    k_attn<<<blks((long long)NNODES * 4, BLK), BLK, 0, stream>>>(xp, att_src, att_dst, a_s, a_d, NNODES * 4);

    // 4) init accumulators
    k_fill<<<blks(NNODES * 4, BLK), BLK, 0, stream>>>(mx, -__builtin_inff(), NNODES * 4);
    k_fill<<<blks(NNODES * 4, BLK), BLK, 0, stream>>>(denom, 0.f, NNODES * 4);
    k_fill<<<blks(NNODES, BLK), BLK, 0, stream>>>(deg, 0.f, NNODES);
    k_fill<<<blks((long long)NNODES * FEAT, BLK), BLK, 0, stream>>>(y_gat, 0.f, NNODES * FEAT);
    k_fill<<<blks((long long)NNODES * FEAT, BLK), BLK, 0, stream>>>(out, 0.f, NNODES * FEAT);

    // 5) segment softmax + GAT aggregation (self-loops folded in, idx >= E)
    k_edge_max  <<<blks(ETOT, BLK), BLK, 0, stream>>>(src, dst, a_s, a_d, mx);
    k_edge_denom<<<blks(ETOT, BLK), BLK, 0, stream>>>(src, dst, ew, a_s, a_d, mx, denom, deg);
    k_gat_scatter<<<blks((long long)ETOT * 32, BLK), BLK, 0, stream>>>(src, dst, a_s, a_d, mx, denom, xp, y_gat);

    // 6) h = y_gat @ W_gcn  (WMMA)
    k_cvt_f16<<<blks((long long)NNODES * FEAT / 4, BLK), BLK, 0, stream>>>(y_gat, ygh, NNODES * FEAT / 4);
    k_gemm_wmma<<<NNODES / 16, 256, 0, stream>>>(ygh, WgcnBp, hbuf, NNODES);

    // 7) GCN normalization + scatter
    k_dinv<<<blks(NNODES, BLK), BLK, 0, stream>>>(deg, NNODES);
    k_gcn_scatter<<<blks((long long)ETOT * 32, BLK), BLK, 0, stream>>>(src, dst, ew, deg, hbuf, out);
}
